// AutoregressiveEdgeDecoder_48619029791018
// MI455X (gfx1250) — matrix operand order repaired and checked by reading, not compiled, not run on hardware
//
#include <hip/hip_runtime.h>

typedef __attribute__((ext_vector_type(16))) _Float16 v16h;
typedef __attribute__((ext_vector_type(8)))  _Float16 v8h;
typedef __attribute__((ext_vector_type(8)))  float    v8f;

#define NN 96   // nodes
#define DH 64   // hidden dim

// ---------------------------------------------------------------------------
// Kernel 0: pair-independent precompute  ZW = z @ W1[:64,:]   (96 x 64, fp32)
// ---------------------------------------------------------------------------
__global__ __launch_bounds__(64)
void zw_kernel(const float* __restrict__ z, const float* __restrict__ W1,
               float* __restrict__ ZW) {
  const int m = blockIdx.x;   // 0..95
  const int n = threadIdx.x;  // 0..63
  float acc = 0.f;
#pragma unroll 8
  for (int k = 0; k < 64; ++k) acc += z[m * 64 + k] * W1[k * 64 + n];
  ZW[m * 64 + n] = acc;
}

// ---------------------------------------------------------------------------
// Kernel 1: one block per pair (i,j). 128 threads = 4 waves (wave32).
// conv1 (pa @ x) via v_wmma_f32_16x16x32_f16 with hi/lo f16 split (pa is
// binary -> exact in f16; hi+lo split recovers ~fp32 accuracy, fp32 accum).
// Each wave owns one 16-column strip: B fragments loaded once, held in regs.
// ---------------------------------------------------------------------------
__global__ __launch_bounds__(128)
void pair_kernel(const float* __restrict__ adj, const float* __restrict__ W1,
                 const float* __restrict__ W2, const float* __restrict__ ZW,
                 float* __restrict__ out) {
  __shared__ __align__(16) _Float16 pa[NN][NN];     // binary mask, exact in f16
  __shared__ __align__(16) _Float16 xhiT[DH][NN];   // (deg*x) hi part, transposed
  __shared__ __align__(16) _Float16 xloT[DH][NN];   // (deg*x) lo part, transposed
  __shared__ float degs[NN];
  __shared__ float tacc[NN];                        // t = relu(conv1) @ W2
  __shared__ float w2s[DH];
  __shared__ float red[64];

  const int p = blockIdx.x;
  const int i = p / NN;
  const int j = p % NN;
  const int u = (i > j) ? i : j;
  const int l = (i > j) ? j : i;
  const int tid  = threadIdx.x;
  const int wave = tid >> 5;
  const int lane = tid & 31;
  const int l16  = lane & 15;
  const int lhalf = lane >> 4;  // 0 or 1

  // ---- stage 1: build symmetrized masked adjacency (binary) + init ----
  for (int e = tid; e < NN * NN; e += 128) {
    const int r = e / NN, c = e % NN;
    // keep | keep^T from the reference (already symmetric):
    const bool keep = (r < u && c < u) || (r == u && c < l) || (c == u && r < l);
    float v = keep ? adj[e] : 0.f;
    if (r == c) v = 1.f;                 // max with identity
    pa[r][c] = (_Float16)v;              // 0.0 or 1.0 -> exact in f16
  }
  if (tid < DH) w2s[tid] = W2[tid];
  if (tid < NN) tacc[tid] = 0.f;
  __syncthreads();

  // ---- stage 2: degree scaling ----
  if (tid < NN) {
    float s = 0.f;
    for (int r = 0; r < NN; ++r) s += (float)pa[r][tid];
    degs[tid] = rsqrtf(fmaxf(s, 1.0f));
  }
  __syncthreads();

  // ---- stage 3: x = deg * (ZW + e_i*W1[64] + e_j*W1[65]); split hi/lo f16 ----
  for (int e = tid; e < NN * DH; e += 128) {
    const int m = e / DH, n = e % DH;
    float x = ZW[e];
    if (m == i) x += W1[64 * DH + n];
    if (m == j) x += W1[65 * DH + n];
    x *= degs[m];
    const _Float16 hi = (_Float16)x;
    const _Float16 lo = (_Float16)(x - (float)hi);
    xhiT[n][m] = hi;
    xloT[n][m] = lo;
  }
  __syncthreads();

  // ---- stage 4: conv1 = pa @ x via WMMA; fold relu/deg/W2 into tacc[] ----
  // 96x64 output = 6x4 tiles of 16x16; wave w owns column strip nt = w.
  const int nt = wave;
  // Load all B fragments for this strip once (3 k-steps x {hi,lo}).
  const v8h* bhrow = (const v8h*)(&xhiT[nt * 16 + l16][0]);   // stride 192B
  const v8h* blrow = (const v8h*)(&xloT[nt * 16 + l16][0]);
  v16h bh[3], bl[3];
#pragma unroll
  for (int kk = 0; kk < 3; ++kk) {
    const int bb8 = kk * 4 + lhalf;                            // v8h units
    v8h b0 = bhrow[bb8], b1 = bhrow[bb8 + 2];
#pragma unroll
    for (int q = 0; q < 8; ++q) { bh[kk][q] = b0[q]; bh[kk][q + 8] = b1[q]; }
    b0 = blrow[bb8]; b1 = blrow[bb8 + 2];
#pragma unroll
    for (int q = 0; q < 8; ++q) { bl[kk][q] = b0[q]; bl[kk][q + 8] = b1[q]; }
  }
  const int ncol = nt * 16 + l16;
  const float w2n = w2s[ncol];

#pragma unroll
  for (int mt = 0; mt < 6; ++mt) {
    v8f acc = {};
    const v8h* arow = (const v8h*)(&pa[mt * 16 + l16][0]);     // stride 192B
#pragma unroll
    for (int kk = 0; kk < 3; ++kk) {
      // A frag: documented 16-bit 16x32 layout (K chunks {b..b+7},{b+16..b+23})
      const int kb8 = kk * 4 + lhalf;
      const v8h a0 = arow[kb8];
      const v8h a1 = arow[kb8 + 2];
      v16h af;
#pragma unroll
      for (int q = 0; q < 8; ++q) { af[q] = a0[q]; af[q + 8] = a1[q]; }
      acc = __builtin_amdgcn_wmma_f32_16x16x32_f16(false, af, false, bh[kk],
                                                   (short)0, acc, false, false);
      acc = __builtin_amdgcn_wmma_f32_16x16x32_f16(false, af, false, bl[kk],
                                                   (short)0, acc, false, false);
    }
    // D layout: lane column n = nt*16+l16; row m = mt*16 + v + 8*lhalf
#pragma unroll
    for (int v = 0; v < 8; ++v) {
      const int m = mt * 16 + v + 8 * lhalf;
      const float hval = fmaxf(degs[m] * acc[v], 0.f) * w2n;   // relu(conv1)@W2
      atomicAdd(&tacc[m], hval);
    }
  }
  __syncthreads();

  // ---- stage 5: s = deg * t ----
  if (tid < NN) tacc[tid] *= degs[tid];
  __syncthreads();

  // ---- stage 6: h2[row] = deg[row] * (pa[row] . s), only rows i and j ----
  if (wave < 2) {
    const int row = (wave == 0) ? i : j;
    float s = 0.f;
    for (int c = lane; c < NN; c += 32) s += (float)pa[row][c] * tacc[c];
    red[wave * 32 + lane] = s;
  }
  __syncthreads();
  if (tid == 0) {
    float si = 0.f, sj = 0.f;
    for (int q = 0; q < 32; ++q) { si += red[q]; sj += red[32 + q]; }
    out[p] = degs[i] * si + degs[j] * sj;
  }
}

// ---------------------------------------------------------------------------
extern "C" void kernel_launch(void* const* d_in, const int* in_sizes, int n_in,
                              void* d_out, int out_size, void* d_ws, size_t ws_size,
                              hipStream_t stream) {
  const float* z   = (const float*)d_in[0];  // (96,64)
  const float* adj = (const float*)d_in[1];  // (96,96)
  const float* W1  = (const float*)d_in[2];  // (66,64)
  const float* W2  = (const float*)d_in[3];  // (64,1)
  float* out = (float*)d_out;                // 9216
  float* ZW  = (float*)d_ws;                 // 96*64 fp32 scratch

  zw_kernel<<<NN, DH, 0, stream>>>(z, W1, ZW);
  pair_kernel<<<NN * NN, 128, 0, stream>>>(adj, W1, W2, ZW, out);
}
